// GAT_21509196218426
// MI455X (gfx1250) — matrix operand order repaired and checked by reading, not compiled, not run on hardware
//
#include <hip/hip_runtime.h>
#include <math.h>

// ---------------- problem constants (match reference) ----------------
#define N_NODES 10000
#define MP      10048            // N_NODES padded to multiple of 64
#define N_EDGES 80000            // raw edges
#define E_TOT   (N_EDGES + N_NODES) // + self loops
#define IN_FEAT 200
#define KP1     224              // IN_FEAT padded to multiple of 32
#define KT1     (KP1/32)         // 7 K-tiles
#define H1      8
#define C1      400
#define HC1     (H1*C1)          // 3200
#define KT2     (HC1/32)         // 100 K-tiles
#define C2      200
#define NB2     256              // C2 padded to multiple of 128

// ---------------- WMMA types ----------------
typedef __bf16 bf16_t;
typedef bf16_t v16bf __attribute__((ext_vector_type(16)));
typedef float  v8f   __attribute__((ext_vector_type(8)));

union Frag { v16bf v; uint4 q[2]; unsigned short u[16]; };

__device__ inline unsigned short f32_to_bf16(float f) {
    unsigned int u = __float_as_uint(f);
    unsigned int r = u + 0x7FFFu + ((u >> 16) & 1u);
    return (unsigned short)(r >> 16);
}
__device__ inline float bf16_to_f32(unsigned short h) {
    return __uint_as_float(((unsigned int)h) << 16);
}
// monotone float<->uint key for atomic max on floats
__device__ inline unsigned int fkey(float f) {
    unsigned int u = __float_as_uint(f);
    return (u & 0x80000000u) ? ~u : (u | 0x80000000u);
}
__device__ inline float fdecode(unsigned int k) {
    unsigned int u = (k & 0x80000000u) ? (k ^ 0x80000000u) : ~k;
    return __uint_as_float(u);
}
__device__ inline void edge_sd(const int* ei, int e, int& s, int& d) {
    if (e < N_EDGES) { s = ei[e]; d = ei[N_EDGES + e]; }
    else             { s = e - N_EDGES; d = e - N_EDGES; }
}
__device__ inline float ldval(const float* p)          { return *p; }
__device__ inline float ldval(const unsigned short* p) { return bf16_to_f32(*p); }

// CDNA5 async global->LDS (ASYNCcnt path). Generic LDS pointer low 32 bits are
// the LDS byte address (aperture rule).
__device__ __forceinline__ void async_cp16(unsigned lds_addr, const void* g) {
    asm volatile("global_load_async_to_lds_b128 %0, %1, off"
                 :: "v"(lds_addr), "v"((unsigned long long)(uintptr_t)g)
                 : "memory");
}
template<int N>
__device__ __forceinline__ void async_wait() {
    asm volatile("s_wait_asynccnt %0" :: "i"(N) : "memory");
}

// ---------------- utility kernels ----------------
__global__ void k_zero_f32(float* p, long n) {
    long i = (long)blockIdx.x * blockDim.x + threadIdx.x;
    if (i < n) p[i] = 0.0f;
}

// fp32 -> bf16 with zero padding to (drows, dcols)
__global__ void k_cvt_pad_bf16(const float* __restrict__ src, unsigned short* __restrict__ dst,
                               int srows, int scols, int drows, int dcols) {
    long i = (long)blockIdx.x * blockDim.x + threadIdx.x;
    long tot = (long)drows * dcols;
    if (i >= tot) return;
    int r = (int)(i / dcols), c = (int)(i % dcols);
    float v = (r < srows && c < scols) ? src[(long)r * scols + c] : 0.0f;
    dst[i] = f32_to_bf16(v);
}

// pack weights into per-K-tile column-major bf16: dst[((t*NB)+n)*32 + kk] = W[(t*32+kk), n]
__global__ void k_pack_w_bf16(const float* __restrict__ src, unsigned short* __restrict__ dst,
                              int srcK, int srcN, int KT, int NB) {
    long i = (long)blockIdx.x * blockDim.x + threadIdx.x;
    long tot = (long)KT * NB * 32;
    if (i >= tot) return;
    int kk = (int)(i & 31);
    long r = i >> 5;
    int n = (int)(r % NB);
    int t = (int)(r / NB);
    int k = t * 32 + kk;
    float v = (k < srcK && n < srcN) ? src[(long)k * srcN + n] : 0.0f;
    dst[i] = f32_to_bf16(v);
}

// ---------------- tiled bf16 WMMA GEMM: C[MP,NB] = A[MP,K] x Bp ----------------
// A: row-major bf16 [MP x K] (K mult 32, MP mult 64, zero padded -> NO guards)
// Bp: packed per-K-tile column-major [K/32][NB][32]   (NB mult of 128)
// Block tile 64x128; 8 waves in 2(M) x 4(N); each wave owns a 32x32 tile
// = 4 v_wmma_f32_16x16x32_bf16 per k-step. Double-buffered LDS staged with
// global_load_async_to_lds_b128 (3 copies/thread/tile), threshold waits.
template<bool BF16OUT>
__global__ __launch_bounds__(256)
void k_gemm_wmma(const unsigned short* __restrict__ A, const unsigned short* __restrict__ Bp,
                 void* __restrict__ Cp, int K, int lda, int NB, int ldc) {
    __shared__ __align__(16) unsigned short As[2][64 * 32];    // row-major, 64B rows
    __shared__ __align__(16) unsigned short Bs[2][128 * 32];   // col(n)-major, 64B cols

    const int tid  = threadIdx.x;
    const int lane = tid & 31;
    const int wave = tid >> 5;
    const int mw   = wave >> 2;   // 0..1  (M strip of 32)
    const int nw   = wave & 3;    // 0..3  (N strip of 32)
    const int half = lane >> 4;   // 0..1
    const int l16  = lane & 15;

    const int row0 = blockIdx.y * 64;
    const int col0 = blockIdx.x * 128;

    // staging coordinates: A tile 4KB (16B/thread), B tile 8KB (2x16B/thread)
    const int ar  = tid >> 2;          // 0..63
    const int ac8 = (tid & 3) * 8;     // 0,8,16,24
    const unsigned ldsA0 = (unsigned)(uintptr_t)&As[0][0] + (unsigned)(ar * 32 + ac8) * 2u;
    const unsigned ldsB0 = (unsigned)(uintptr_t)&Bs[0][0] + (unsigned)tid * 32u;
    const unsigned bufStrideA = 64 * 32 * 2u;
    const unsigned bufStrideB = 128 * 32 * 2u;

    const unsigned short* gA = A + (size_t)(row0 + ar) * lda + ac8;
    const unsigned short* gB = Bp + (size_t)col0 * 32 + (size_t)tid * 16;
    const size_t bTileStride = (size_t)NB * 32;

    v8f acc00 = {0.f,0.f,0.f,0.f,0.f,0.f,0.f,0.f};
    v8f acc01 = acc00, acc10 = acc00, acc11 = acc00;

    const int T = K >> 5;
    // prologue: stage tile 0 into buffer 0
    async_cp16(ldsA0, gA);
    async_cp16(ldsB0, gB);
    async_cp16(ldsB0 + 16u, gB + 8);

    for (int t = 0; t < T; ++t) {
        const int cur = t & 1;
        if (t + 1 < T) {   // stage next tile into other buffer, then drain current
            const int nb = cur ^ 1;
            async_cp16(ldsA0 + nb * bufStrideA, gA + (size_t)(t + 1) * 32);
            async_cp16(ldsB0 + nb * bufStrideB, gB + (size_t)(t + 1) * bTileStride);
            async_cp16(ldsB0 + nb * bufStrideB + 16u, gB + (size_t)(t + 1) * bTileStride + 8);
            if (t + 2 < T) {   // global_prefetch_b8 two tiles ahead
                __builtin_prefetch(gA + (size_t)(t + 2) * 32, 0, 3);
                __builtin_prefetch(gB + (size_t)(t + 2) * bTileStride, 0, 3);
            }
            async_wait<3>();   // only tile-t copies must be complete
        } else {
            async_wait<0>();
        }
        __syncthreads();

        // A fragments: rows mw*32 + {0..15, 16..31}; two contiguous 16B chunks/lane
        const uint4* ar0 = (const uint4*)(&As[cur][0] + (mw * 32 + l16) * 32);
        const uint4* ar1 = (const uint4*)(&As[cur][0] + (mw * 32 + 16 + l16) * 32);
        Frag fa0; fa0.q[0] = ar0[half]; fa0.q[1] = ar0[2 + half];
        Frag fa1; fa1.q[0] = ar1[half]; fa1.q[1] = ar1[2 + half];
        // B fragments: cols nw*32 + {0..15, 16..31}; two contiguous 16B chunks/lane
        const uint4* br0 = (const uint4*)(&Bs[cur][0] + (nw * 32 + l16) * 32);
        const uint4* br1 = (const uint4*)(&Bs[cur][0] + (nw * 32 + 16 + l16) * 32);
        Frag fb0; fb0.q[0] = br0[half]; fb0.q[1] = br0[2 + half];
        Frag fb1; fb1.q[0] = br1[half]; fb1.q[1] = br1[2 + half];

        acc00 = __builtin_amdgcn_wmma_f32_16x16x32_bf16(false, fa0.v, false, fb0.v,
                                                        (short)0, acc00, false, false);
        acc01 = __builtin_amdgcn_wmma_f32_16x16x32_bf16(false, fa0.v, false, fb1.v,
                                                        (short)0, acc01, false, false);
        acc10 = __builtin_amdgcn_wmma_f32_16x16x32_bf16(false, fa1.v, false, fb0.v,
                                                        (short)0, acc10, false, false);
        acc11 = __builtin_amdgcn_wmma_f32_16x16x32_bf16(false, fa1.v, false, fb1.v,
                                                        (short)0, acc11, false, false);
        __syncthreads();
    }

    // store: C/D layout — VGPR r: lanes0-15 -> M=r, lanes16-31 -> M=8+r; N=lane%16
    #pragma unroll
    for (int r = 0; r < 8; ++r) {
        int cm0 = row0 + mw * 32 + half * 8 + r;
        int cm1 = cm0 + 16;
        int cn0 = col0 + nw * 32 + l16;
        int cn1 = cn0 + 16;
        if (BF16OUT) {
            unsigned short* C = (unsigned short*)Cp;
            C[(size_t)cm0 * ldc + cn0] = f32_to_bf16(acc00[r]);
            C[(size_t)cm0 * ldc + cn1] = f32_to_bf16(acc01[r]);
            C[(size_t)cm1 * ldc + cn0] = f32_to_bf16(acc10[r]);
            C[(size_t)cm1 * ldc + cn1] = f32_to_bf16(acc11[r]);
        } else {
            float* C = (float*)Cp;
            C[(size_t)cm0 * ldc + cn0] = acc00[r];
            C[(size_t)cm0 * ldc + cn1] = acc01[r];
            C[(size_t)cm1 * ldc + cn0] = acc10[r];
            C[(size_t)cm1 * ldc + cn1] = acc11[r];
        }
    }
}

// ---------------- attention logit projections: one wave per (node, head) ----------------
template<typename T>
__global__ __launch_bounds__(256)
void k_alpha_proj(const T* __restrict__ h, const float* __restrict__ asrc,
                  const float* __restrict__ adst, float* __restrict__ als,
                  float* __restrict__ ald, int n, int H, int C, int ld) {
    int wid  = (blockIdx.x * blockDim.x + threadIdx.x) >> 5;
    int lane = threadIdx.x & 31;
    if (wid >= n * H) return;
    int i = wid / H, hh = wid % H;
    float s = 0.f, d = 0.f;
    for (int c = lane; c < C; c += 32) {
        float x = ldval(&h[(long)i * ld + hh * C + c]);
        s += x * asrc[hh * C + c];
        d += x * adst[hh * C + c];
    }
    #pragma unroll
    for (int m = 16; m > 0; m >>= 1) {
        s += __shfl_xor(s, m, 32);
        d += __shfl_xor(d, m, 32);
    }
    if (lane == 0) { als[wid] = s; ald[wid] = d; }
}

// ---------------- edge pass 1: leaky-relu logits + segment max ----------------
__global__ void k_edge_logit_max(const int* __restrict__ ei, const float* __restrict__ als,
                                 const float* __restrict__ ald, float* __restrict__ elog,
                                 unsigned int* __restrict__ kmax, int H) {
    long g = (long)blockIdx.x * blockDim.x + threadIdx.x;
    if (g >= (long)E_TOT * H) return;
    int e = (int)(g / H), hh = (int)(g % H);
    int s, d; edge_sd(ei, e, s, d);
    float x = als[(long)s * H + hh] + ald[(long)d * H + hh];
    x = (x > 0.f) ? x : 0.2f * x;
    elog[g] = x;
    atomicMax(&kmax[(long)d * H + hh], fkey(x));
}

// ---------------- edge pass 2: exp + segment sum ----------------
__global__ void k_edge_exp_den(const int* __restrict__ ei, const float* __restrict__ elog,
                               const unsigned int* __restrict__ kmax, float* __restrict__ ew,
                               float* __restrict__ den, int H) {
    long g = (long)blockIdx.x * blockDim.x + threadIdx.x;
    if (g >= (long)E_TOT * H) return;
    int e = (int)(g / H), hh = (int)(g % H);
    int s, d; edge_sd(ei, e, s, d);
    (void)s;
    float w = __expf(elog[g] - fdecode(kmax[(long)d * H + hh]));
    ew[g] = w;
    atomicAdd(&den[(long)d * H + hh], w);
}

// ---------------- edge pass 3: scatter-add messages (one block per edge) ----------------
template<typename T>
__global__ __launch_bounds__(256)
void k_aggregate(const int* __restrict__ ei, const T* __restrict__ hsrc,
                 const float* __restrict__ ew, const float* __restrict__ den,
                 float* __restrict__ out, int H, int C, int ld_src, int ld_out) {
    __shared__ float alpha_s[8];
    int e = blockIdx.x;
    int s, d; edge_sd(ei, e, s, d);
    if (threadIdx.x < H)
        alpha_s[threadIdx.x] = ew[(long)e * H + threadIdx.x] / den[(long)d * H + threadIdx.x];
    __syncthreads();
    int HC = H * C;
    for (int c = threadIdx.x; c < HC; c += blockDim.x) {
        int hh = c / C;
        float v = ldval(&hsrc[(long)s * ld_src + c]) * alpha_s[hh];
        atomicAdd(&out[(long)d * ld_out + c], v);
    }
}

// ---------------- bias + ELU + bf16 convert (layer1 -> layer2 input) ----------------
__global__ void k_bias_elu_bf16(const float* __restrict__ in, const float* __restrict__ b,
                                unsigned short* __restrict__ outb, long n, int cols) {
    long i = (long)blockIdx.x * blockDim.x + threadIdx.x;
    if (i >= n) return;
    float v = in[i] + b[i % cols];
    v = (v > 0.f) ? v : (__expf(v) - 1.f);
    outb[i] = f32_to_bf16(v);
}

__global__ void k_bias_add(float* __restrict__ out, const float* __restrict__ b, long n, int cols) {
    long i = (long)blockIdx.x * blockDim.x + threadIdx.x;
    if (i >= n) return;
    out[i] += b[i % cols];
}

// ---------------- host-side orchestration ----------------
static inline size_t align256(size_t x) { return (x + 255) & ~(size_t)255; }

extern "C" void kernel_launch(void* const* d_in, const int* in_sizes, int n_in,
                              void* d_out, int out_size, void* d_ws, size_t ws_size,
                              hipStream_t stream) {
    const float* x      = (const float*)d_in[0];
    const int*   ei     = (const int*)d_in[1];
    const float* W1     = (const float*)d_in[2];
    const float* a_src1 = (const float*)d_in[3];
    const float* a_dst1 = (const float*)d_in[4];
    const float* b1     = (const float*)d_in[5];
    const float* W2     = (const float*)d_in[6];
    const float* a_src2 = (const float*)d_in[7];
    const float* a_dst2 = (const float*)d_in[8];
    const float* b2     = (const float*)d_in[9];
    float* out = (float*)d_out;

    // workspace carve-up (all GEMM operands padded: M->MP, K->mult32, N->mult128)
    char* p = (char*)d_ws; size_t off = 0;
    auto carve = [&](size_t bytes) { char* r = p + off; off = align256(off + bytes); return (void*)r; };
    unsigned short* xb   = (unsigned short*)carve((size_t)MP * KP1 * 2);
    unsigned short* W1p  = (unsigned short*)carve((size_t)KT1 * HC1 * 32 * 2);
    unsigned short* h1b  = (unsigned short*)carve((size_t)MP * HC1 * 2);
    float*          als1 = (float*)carve((size_t)N_NODES * H1 * 4);
    float*          ald1 = (float*)carve((size_t)N_NODES * H1 * 4);
    float*          elog1= (float*)carve((size_t)E_TOT * H1 * 4);
    float*          ew1  = (float*)carve((size_t)E_TOT * H1 * 4);
    unsigned int*   kmx1 = (unsigned int*)carve((size_t)N_NODES * H1 * 4);
    float*          den1 = (float*)carve((size_t)N_NODES * H1 * 4);
    float*          out1 = (float*)carve((size_t)N_NODES * HC1 * 4);
    unsigned short* x2b  = (unsigned short*)carve((size_t)MP * HC1 * 2);
    unsigned short* W2p  = (unsigned short*)carve((size_t)KT2 * NB2 * 32 * 2);
    float*          h2   = (float*)carve((size_t)MP * NB2 * 4);
    float*          als2 = (float*)carve((size_t)N_NODES * 4);
    float*          ald2 = (float*)carve((size_t)N_NODES * 4);
    float*          elog2= (float*)carve((size_t)E_TOT * 4);
    float*          ew2  = (float*)carve((size_t)E_TOT * 4);
    unsigned int*   kmx2 = (unsigned int*)carve((size_t)N_NODES * 4);
    float*          den2 = (float*)carve((size_t)N_NODES * 4);
    (void)ws_size; (void)in_sizes; (void)n_in; (void)out_size;

    auto g1 = [](long n) { return dim3((unsigned)((n + 255) / 256)); };

    // zero-init accumulators / keys / x2b pad rows (deterministic every call)
    k_zero_f32<<<g1((long)N_NODES * HC1), 256, 0, stream>>>(out1, (long)N_NODES * HC1);
    k_zero_f32<<<g1((long)N_NODES * H1), 256, 0, stream>>>((float*)kmx1, (long)N_NODES * H1);
    k_zero_f32<<<g1((long)N_NODES * H1), 256, 0, stream>>>(den1, (long)N_NODES * H1);
    k_zero_f32<<<g1(N_NODES), 256, 0, stream>>>((float*)kmx2, N_NODES);
    k_zero_f32<<<g1(N_NODES), 256, 0, stream>>>(den2, N_NODES);
    k_zero_f32<<<g1((long)N_NODES * C2), 256, 0, stream>>>(out, (long)N_NODES * C2);
    k_zero_f32<<<g1((long)(MP - N_NODES) * HC1 / 2), 256, 0, stream>>>(
        (float*)(x2b + (size_t)N_NODES * HC1), (long)(MP - N_NODES) * HC1 / 2);

    // bf16 conversions / packing
    k_cvt_pad_bf16<<<g1((long)MP * KP1), 256, 0, stream>>>(x, xb, N_NODES, IN_FEAT, MP, KP1);
    k_pack_w_bf16<<<g1((long)KT1 * HC1 * 32), 256, 0, stream>>>(W1, W1p, IN_FEAT, HC1, KT1, HC1);
    k_pack_w_bf16<<<g1((long)KT2 * NB2 * 32), 256, 0, stream>>>(W2, W2p, HC1, C2, KT2, NB2);

    // ---- layer 1 ----
    {
        dim3 grid(HC1 / 128, MP / 64);
        k_gemm_wmma<true><<<grid, 256, 0, stream>>>(xb, W1p, (void*)h1b, KP1, KP1, HC1, HC1);
    }
    k_alpha_proj<unsigned short><<<g1((long)N_NODES * H1 * 32), 256, 0, stream>>>(
        h1b, a_src1, a_dst1, als1, ald1, N_NODES, H1, C1, HC1);
    k_edge_logit_max<<<g1((long)E_TOT * H1), 256, 0, stream>>>(ei, als1, ald1, elog1, kmx1, H1);
    k_edge_exp_den<<<g1((long)E_TOT * H1), 256, 0, stream>>>(ei, elog1, kmx1, ew1, den1, H1);
    k_aggregate<unsigned short><<<E_TOT, 256, 0, stream>>>(ei, h1b, ew1, den1, out1,
                                                           H1, C1, HC1, HC1);
    k_bias_elu_bf16<<<g1((long)N_NODES * HC1), 256, 0, stream>>>(out1, b1, x2b,
                                                                 (long)N_NODES * HC1, HC1);

    // ---- layer 2 ----
    {
        dim3 grid(NB2 / 128, MP / 64);
        k_gemm_wmma<false><<<grid, 256, 0, stream>>>(x2b, W2p, (void*)h2, HC1, HC1, NB2, NB2);
    }
    k_alpha_proj<float><<<g1((long)N_NODES * 32), 256, 0, stream>>>(
        h2, a_src2, a_dst2, als2, ald2, N_NODES, 1, C2, NB2);
    k_edge_logit_max<<<g1((long)E_TOT), 256, 0, stream>>>(ei, als2, ald2, elog2, kmx2, 1);
    k_edge_exp_den<<<g1((long)E_TOT), 256, 0, stream>>>(ei, elog2, kmx2, ew2, den2, 1);
    k_aggregate<float><<<E_TOT, 256, 0, stream>>>(ei, h2, ew2, den2, out, 1, C2, NB2, C2);
    k_bias_add<<<g1((long)N_NODES * C2), 256, 0, stream>>>(out, b2, (long)N_NODES * C2, C2);
}
